// LSTM_20298015441238
// MI455X (gfx1250) — matrix operand order repaired and checked by reading, not compiled, not run on hardware
//
#include <hip/hip_runtime.h>
#include <hip/hip_bf16.h>

typedef __attribute__((ext_vector_type(16))) __bf16 v16bf;
typedef __attribute__((ext_vector_type(8)))  float  v8f;
typedef __attribute__((ext_vector_type(4)))  unsigned int u32x4;

#define SEQ    2048
#define BATCH  64
#define INDIM  256
#define HID    512
#define G4     2048      // 4*HID
#define NT     128       // G4/16 column tiles
#define KT0    24        // (INDIM+HID)/32
#define KT1    32        // (HID+HID)/32

// ws layout (bytes)
#define PK0_OFF   0
#define PK0_BYTES (KT0*NT*512*2)           // 3,145,728
#define PK1_OFF   (PK0_OFF + PK0_BYTES)
#define PK1_BYTES (KT1*NT*512*2)           // 4,194,304
#define B0_OFF    (PK1_OFF + PK1_BYTES)
#define B1_OFF    (B0_OFF + G4*4)
#define C_OFF     (B1_OFF + G4*4)          // 2*64*512 f32 = 256KB

// LDS layout (bytes)
#define L_XB    0                           // 64*256 bf16 = 32KB
#define L_H0A   32768                       // 64*512 bf16 = 64KB
#define L_H0B   (32768+65536)
#define L_H1A   163840
#define L_H1B   229376
#define L_HD    294912                      // 256 f32
#define LDS_BYTES 295936

__device__ __forceinline__ float sigf(float x)   { return 1.0f / (1.0f + __expf(-x)); }
__device__ __forceinline__ float tanh_f(float x) { float e = __expf(-2.0f * x); return (1.0f - e) / (1.0f + e); }

// ---------------------------------------------------------------------------
// Pack fp32 weights into per-lane WMMA-B bf16 tiles (contiguous k per lane half, same order as A) + fused biases.
// Tile (nt,kt) = 1KB at (nt*KT+kt)*512 elems; within: lane*16+kk,
// lane=(n%16)+16*khalf, covers B[k][n]=W[n][k], k = kt*32 + khalf*16 + kk.
// ---------------------------------------------------------------------------
__global__ void lstm_pack(const float* __restrict__ w_ih0, const float* __restrict__ w_hh0,
                          const float* __restrict__ b_ih0, const float* __restrict__ b_hh0,
                          const float* __restrict__ w_ih1, const float* __restrict__ w_hh1,
                          const float* __restrict__ b_ih1, const float* __restrict__ b_hh1,
                          __bf16* __restrict__ pk0, __bf16* __restrict__ pk1,
                          float* __restrict__ bias0, float* __restrict__ bias1)
{
    const int L0E = KT0 * NT * 512;
    const int L1E = KT1 * NT * 512;
    int gid = blockIdx.x * 256 + threadIdx.x;
    if (gid < L0E) {
        int tile = gid >> 9, within = gid & 511;
        int lane = within >> 4, kk = within & 15;
        int nt = tile / KT0, kt = tile % KT0;
        int n = nt * 16 + (lane & 15);
        int k = kt * 32 + (lane >> 4) * 16 + kk;
        float v = (k < INDIM) ? w_ih0[n * INDIM + k] : w_hh0[n * HID + (k - INDIM)];
        pk0[gid] = (__bf16)v;
    } else if (gid < L0E + L1E) {
        int e = gid - L0E;
        int tile = e >> 9, within = e & 511;
        int lane = within >> 4, kk = within & 15;
        int nt = tile >> 5, kt = tile & 31;
        int n = nt * 16 + (lane & 15);
        int k = kt * 32 + (lane >> 4) * 16 + kk;
        float v = (k < HID) ? w_ih1[n * HID + k] : w_hh1[n * HID + (k - HID)];
        pk1[e] = (__bf16)v;
    } else if (gid < L0E + L1E + 2 * G4) {
        int e = gid - L0E - L1E;
        if (e < G4) bias0[e] = b_ih0[e] + b_hh0[e];
        else        bias1[e - G4] = b_ih1[e - G4] + b_hh1[e - G4];
    }
}

// ---------------------------------------------------------------------------
// One layer-step: wave w owns h-columns [16w,16w+16). A = [regA | regB] bf16.
// mt loop kept ROLLED; an opaque integer-offset rebind (address space
// preserving) blocks LICM/CSE of the B-tile loads across mt iterations
// (hoisting them caused catastrophic scratch spills). B re-reads hit WGP$/L2
// via clean global_load_b128. k-loop split in two to avoid per-iteration
// conditional address selects.
// ---------------------------------------------------------------------------
__device__ __forceinline__ void lstm_layer_step(
    const __bf16* __restrict__ regA, int pitchA, int KTA,
    const __bf16* __restrict__ regB, int KT,
    const __bf16* __restrict__ pk,
    float bi, float bf, float bg, float bo,
    float* __restrict__ cstate,          // 64*512 f32 (this layer)
    __bf16* __restrict__ hw,             // 64*512 bf16 LDS write buffer
    int w, int lane, int ln, int kh)
{
    union AF { v16bf v; u32x4 q[2]; };
    const int j = w * 16 + ln;
#pragma unroll 1
    for (int mt = 0; mt < 4; ++mt) {
        // opaque offset: keeps pk in the global address space but makes the
        // per-mt weight loads non-provably-identical -> no hoist, no spill
        unsigned int opq = 0;
        asm volatile("" : "+s"(opq));
        const __bf16* pkl = pk + opq;

        v8f accI = {}, accF = {}, accG = {}, accO = {};
        const __bf16* rowA = regA + (mt * 16 + ln) * pitchA;
        const __bf16* rowB = regB + (mt * 16 + ln) * HID;

#pragma unroll 2
        for (int kt = 0; kt < KTA; ++kt) {              // x / h0-part of A
            AF a;
            a.q[0] = *(const u32x4*)(rowA + kt * 32 + kh * 16);
            a.q[1] = *(const u32x4*)(rowA + kt * 32 + kh * 16 + 8);
#pragma unroll
            for (int G = 0; G < 4; ++G) {
                const u32x4* bp = (const u32x4*)(pkl + (size_t)((G * 32 + w) * KT + kt) * 512) + lane * 2;
                AF b; b.q[0] = bp[0]; b.q[1] = bp[1];
                v8f& acc = (G == 0 ? accI : G == 1 ? accF : G == 2 ? accG : accO);
                acc = __builtin_amdgcn_wmma_f32_16x16x32_bf16(
                        false, a.v, false, b.v, (short)0, acc, false, false);
            }
        }
#pragma unroll 2
        for (int kt = KTA; kt < KT; ++kt) {             // h-part of A
            const int kc = (kt - KTA) * 32;
            AF a;
            a.q[0] = *(const u32x4*)(rowB + kc + kh * 16);
            a.q[1] = *(const u32x4*)(rowB + kc + kh * 16 + 8);
#pragma unroll
            for (int G = 0; G < 4; ++G) {
                const u32x4* bp = (const u32x4*)(pkl + (size_t)((G * 32 + w) * KT + kt) * 512) + lane * 2;
                AF b; b.q[0] = bp[0]; b.q[1] = bp[1];
                v8f& acc = (G == 0 ? accI : G == 1 ? accF : G == 2 ? accG : accO);
                acc = __builtin_amdgcn_wmma_f32_16x16x32_bf16(
                        false, a.v, false, b.v, (short)0, acc, false, false);
            }
        }
        // cell update: lane owns column j, rows mt*16 + r + 8*kh
#pragma unroll
        for (int r = 0; r < 8; ++r) {
            int b = mt * 16 + r + kh * 8;
            float gi = sigf(accI[r] + bi);
            float gf = sigf(accF[r] + bf);
            float gg = tanh_f(accG[r] + bg);
            float go = sigf(accO[r] + bo);
            float c  = cstate[b * HID + j];
            float cn = gf * c + gi * gg;
            cstate[b * HID + j] = cn;
            hw[b * HID + j] = (__bf16)(go * tanh_f(cn));
        }
    }
}

// ---------------------------------------------------------------------------
// Persistent single-workgroup recurrence: 1024 threads = 32 wave32 on one WGP.
// Double-buffered h in LDS, fused input projection (K-extended GEMM), fused
// output head. All WMMA regions are branch-uniform (EXEC all ones).
// ---------------------------------------------------------------------------
__global__ void __launch_bounds__(1024, 1)
lstm_seq(const float* __restrict__ x, const float* __restrict__ h0in,
         const float* __restrict__ c0in,
         const __bf16* __restrict__ pk0, const __bf16* __restrict__ pk1,
         const float* __restrict__ bias0, const float* __restrict__ bias1,
         float* __restrict__ c_ws,
         const float* __restrict__ w_out, const float* __restrict__ b_out,
         float* __restrict__ out)
{
    extern __shared__ char smem[];
    char* sm = smem;
    __bf16* xb = (__bf16*)(sm + L_XB);
    float*  hd = (float*)(sm + L_HD);

    const int tid  = threadIdx.x;
    const int w    = tid >> 5;
    const int lane = tid & 31;
    const int ln   = lane & 15;
    const int kh   = lane >> 4;

    // init h (bf16 in LDS, ping-pong slot 0) and c (f32 scratch)
    {
        __bf16* h0i = (__bf16*)(sm + L_H0A);
        __bf16* h1i = (__bf16*)(sm + L_H1A);
        for (int i = tid; i < BATCH * HID; i += 1024) {
            h0i[i] = (__bf16)h0in[i];
            h1i[i] = (__bf16)h0in[BATCH * HID + i];
            c_ws[i]               = c0in[i];
            c_ws[BATCH * HID + i] = c0in[BATCH * HID + i];
        }
    }

    // per-lane fused biases (column j, per gate, per layer) — constant over t
    const int j = w * 16 + ln;
    const float bi0 = bias0[j],         bf0 = bias0[HID + j];
    const float bg0 = bias0[2*HID + j], bo0 = bias0[3*HID + j];
    const float bi1 = bias1[j],         bf1 = bias1[HID + j];
    const float bg1 = bias1[2*HID + j], bo1 = bias1[3*HID + j];

    __syncthreads();

    for (int t = 0; t < SEQ; ++t) {
        const int p = t & 1;
        __bf16* h0r = (__bf16*)(sm + (p ? L_H0B : L_H0A));
        __bf16* h0w = (__bf16*)(sm + (p ? L_H0A : L_H0B));
        __bf16* h1r = (__bf16*)(sm + (p ? L_H1B : L_H1A));
        __bf16* h1w = (__bf16*)(sm + (p ? L_H1A : L_H1B));

        // stage x_t -> LDS (bf16); prefetch next step's tile
        const float* xt = x + (size_t)t * BATCH * INDIM;
        for (int i = tid; i < BATCH * INDIM; i += 1024) xb[i] = (__bf16)xt[i];
        if (tid < 256 && t + 1 < SEQ)
            __builtin_prefetch(x + (size_t)(t + 1) * BATCH * INDIM + tid * 64, 0, 1);
        __syncthreads();

        // layer 0: A = [x_t (K=256) | h0_prev (K=512)]
        lstm_layer_step(xb, INDIM, 8, h0r, KT0, pk0,
                        bi0, bf0, bg0, bo0, c_ws, h0w, w, lane, ln, kh);
        __syncthreads();

        // layer 1: A = [h0_new (K=512) | h1_prev (K=512)]
        lstm_layer_step(h0w, HID, 16, h1r, KT1, pk1,
                        bi1, bf1, bg1, bo1, c_ws + BATCH * HID, h1w, w, lane, ln, kh);
        __syncthreads();

        // fused head: out[t,b] = sigmoid(h1 . w_out + b_out)
        if (tid < 256) {
            int b = tid >> 2, q = tid & 3;
            float s = 0.0f;
            for (int jj = q * 128; jj < q * 128 + 128; ++jj)
                s += (float)h1w[b * HID + jj] * w_out[jj];
            hd[tid] = s;
        }
        __syncthreads();
        if (tid < BATCH) {
            float s = hd[tid * 4] + hd[tid * 4 + 1] + hd[tid * 4 + 2] + hd[tid * 4 + 3] + b_out[0];
            out[t * BATCH + tid] = sigf(s);
        }
        __syncthreads();
    }
}

extern "C" void kernel_launch(void* const* d_in, const int* in_sizes, int n_in,
                              void* d_out, int out_size, void* d_ws, size_t ws_size,
                              hipStream_t stream) {
    const float* x     = (const float*)d_in[0];
    const float* h0in  = (const float*)d_in[1];
    const float* c0in  = (const float*)d_in[2];
    const float* w_ih0 = (const float*)d_in[3];
    const float* w_hh0 = (const float*)d_in[4];
    const float* b_ih0 = (const float*)d_in[5];
    const float* b_hh0 = (const float*)d_in[6];
    const float* w_ih1 = (const float*)d_in[7];
    const float* w_hh1 = (const float*)d_in[8];
    const float* b_ih1 = (const float*)d_in[9];
    const float* b_hh1 = (const float*)d_in[10];
    const float* w_out = (const float*)d_in[11];
    const float* b_out = (const float*)d_in[12];
    float* out = (float*)d_out;

    char* ws = (char*)d_ws;
    __bf16* pk0  = (__bf16*)(ws + PK0_OFF);
    __bf16* pk1  = (__bf16*)(ws + PK1_OFF);
    float* bias0 = (float*)(ws + B0_OFF);
    float* bias1 = (float*)(ws + B1_OFF);
    float* c_ws  = (float*)(ws + C_OFF);

    const int total = KT0 * NT * 512 + KT1 * NT * 512 + 2 * G4;
    lstm_pack<<<(total + 255) / 256, 256, 0, stream>>>(
        w_ih0, w_hh0, b_ih0, b_hh0, w_ih1, w_hh1, b_ih1, b_hh1,
        pk0, pk1, bias0, bias1);

    (void)hipFuncSetAttribute((const void*)lstm_seq,
                              hipFuncAttributeMaxDynamicSharedMemorySize,
                              (int)LDS_BYTES);
    lstm_seq<<<1, 1024, LDS_BYTES, stream>>>(
        x, h0in, c0in, pk0, pk1, bias0, bias1, c_ws, w_out, b_out, out);
}
